// FractalAttention_59236188946842
// MI455X (gfx1250) — matrix operand order repaired and checked by reading, not compiled, not run on hardware
//
#include <hip/hip_runtime.h>
#include <math.h>

typedef unsigned short u16;
typedef __attribute__((ext_vector_type(16))) __bf16 v16bf;
typedef __attribute__((ext_vector_type(8)))  float  v8f;
typedef __attribute__((ext_vector_type(4)))  unsigned int u32x4;
typedef __attribute__((ext_vector_type(4)))  int i32x4;
typedef __attribute__((ext_vector_type(8)))  int i32x8;

#define S_LEN 2048
#define DMODEL 1024
#define NHEAD 16
#define DHEAD 64
#define MROWS 4096   // B*S

static __device__ __forceinline__ __bf16 us2bf(u16 u) {
    union { u16 s; __bf16 b; } cv; cv.s = u; return cv.b;
}
static __device__ __forceinline__ u16 f2b(float f) {
    unsigned int u = __float_as_uint(f);
    u += 0x7FFFu + ((u >> 16) & 1u);   // round-to-nearest-even
    return (u16)(u >> 16);
}
// Raw LDS byte offset of a __shared__ object (LDS aperture keeps the offset in
// the low 32 bits of the generic address).
static __device__ __forceinline__ unsigned lds_off_of(const void* p) {
    return (unsigned)(uintptr_t)p;
}

// ---------------------------------------------------------------------------
// Tensor Data Mover: DMA a 2-D tile of bf16 (tile_w x tile_h, row stride in
// elements) from global memory into LDS.  D# packing per CDNA5 ISA ch.8.
// Issue from ONE wave only; completion via s_wait_tensorcnt.
// ---------------------------------------------------------------------------
static __device__ __forceinline__ void tdm_load_tile_2d(
    unsigned lds_off, const u16* gptr,
    unsigned tile_w, unsigned tile_h, unsigned long long row_stride_elems)
{
    unsigned long long ga = (unsigned long long)(uintptr_t)gptr;
    u32x4 g0;
    g0[0] = 1u;                                                   // count=1 (valid user D#)
    g0[1] = lds_off;                                              // lds_addr
    g0[2] = (unsigned)ga;                                         // global_addr[31:0]
    g0[3] = (unsigned)((ga >> 32) & 0x01FFFFFFull) | (2u << 30);  // addr[56:32] | type=2
    i32x8 g1;
    g1[0] = 0x10000;                                              // data_size=1 (2B), wg_mask=0
    g1[1] = (int)((tile_w & 0xFFFFu) << 16);                      // tensor_dim0 lo16
    g1[2] = (int)(((tile_w >> 16) & 0xFFFFu) | ((tile_h & 0xFFFFu) << 16)); // dim0 hi | dim1 lo
    g1[3] = (int)(((tile_h >> 16) & 0xFFFFu) | ((tile_w & 0xFFFFu) << 16)); // dim1 hi | tile_dim0
    g1[4] = (int)(tile_h & 0xFFFFu);                              // tile_dim1 (tile_dim2=0)
    g1[5] = (int)(unsigned)(row_stride_elems & 0xFFFFFFFFull);    // dim0_stride lo32
    g1[6] = (int)(unsigned)((row_stride_elems >> 32) & 0xFFFFull);// dim0_stride hi16
    g1[7] = 0;
    i32x4 z4 = {0, 0, 0, 0};
#if __has_include(<hip/amd_detail/amd_gfx1250_TDM.h>)
    i32x8 z8 = {0, 0, 0, 0, 0, 0, 0, 0};
    __builtin_amdgcn_tensor_load_to_lds(g0, g1, z4, z4, z8, 0);
#else
    __builtin_amdgcn_tensor_load_to_lds(g0, g1, z4, z4, 0);
#endif
}

// ---------------------------------------------------------------------------
// fp32 -> bf16 elementwise conversion
// ---------------------------------------------------------------------------
__global__ void fractal_cvt_kernel(const float* __restrict__ in,
                                   u16* __restrict__ out, int n) {
    int i = blockIdx.x * blockDim.x + threadIdx.x;
    if (i < n) out[i] = f2b(in[i]);
}

// ---------------------------------------------------------------------------
// WMMA GEMM: C[4096,1024] = A[4096,1024] * W[1024,1024]^T  (A,W bf16 row-major)
// mode 0: fp32 row-major (M x N);  mode 1: fp32 scattered to (b,h,s,dh).
// Block: 128 threads (4 waves), 64x64 block tile, 16x64 wave tile, K step 32.
// TDM double-buffered LDS staging, software pipelined one tile ahead.
// ---------------------------------------------------------------------------
__global__ __launch_bounds__(128)
void fractal_gemm_kernel(const u16* __restrict__ A, const u16* __restrict__ W,
                         float* __restrict__ Cout, int mode) {
    __shared__ u16 As[2][64 * 32];   // [buf][row][k]
    __shared__ u16 Bs[2][64 * 32];   // [buf][n][k]

    const int tid  = threadIdx.x;
    const int lane = tid & 31;
    const int wave = tid >> 5;
    const int m0 = blockIdx.y * 64;
    const int n0 = blockIdx.x * 64;

    const int row   = lane & 15;
    const int hi    = (lane & 16) ? 1 : 0;
    const int klo_a = hi ? 8 : 0;   // unused (A fragment also uses kb16)
    const int kb16  = hi ? 16 : 0;  // B-fragment K split

    const u16* Abase = A + (size_t)m0 * DMODEL;
    const u16* Wbase = W + (size_t)n0 * DMODEL;

    v8f acc[4] = {};

    const int T = DMODEL / 32;       // 32 K-chunks
    if (tid < 32) {                  // wave 0 drives the TDM (EXEC ignored by TDM)
        tdm_load_tile_2d(lds_off_of(As[0]), Abase, 32, 64, DMODEL);
        tdm_load_tile_2d(lds_off_of(Bs[0]), Wbase, 32, 64, DMODEL);
    }

    for (int t = 0; t < T; ++t) {
        const int cur = t & 1;
        if (tid < 32) {
            if (t + 1 < T) {         // prefetch next tile into the other buffer
                const int k1 = (t + 1) * 32;
                tdm_load_tile_2d(lds_off_of(As[cur ^ 1]), Abase + k1, 32, 64, DMODEL);
                tdm_load_tile_2d(lds_off_of(Bs[cur ^ 1]), Wbase + k1, 32, 64, DMODEL);
                __builtin_amdgcn_s_wait_tensorcnt(2);   // tile t done, t+1 in flight
            } else {
                __builtin_amdgcn_s_wait_tensorcnt(0);
            }
        }
        __syncthreads();             // publish tile t to all waves

        v16bf af;
#pragma unroll
        for (int e = 0; e < 16; ++e) {
            int kk = kb16 + e;
            af[e] = us2bf(As[cur][(wave * 16 + row) * 32 + kk]);
        }
#pragma unroll
        for (int ns = 0; ns < 4; ++ns) {
            v16bf bfr;
#pragma unroll
            for (int e = 0; e < 16; ++e)
                bfr[e] = us2bf(Bs[cur][(ns * 16 + row) * 32 + kb16 + e]);
            acc[ns] = __builtin_amdgcn_wmma_f32_16x16x32_bf16(
                false, af, false, bfr, (short)0, acc[ns], false, false);
        }
        __syncthreads();             // all reads of buf `cur` done before reuse
    }

#pragma unroll
    for (int ns = 0; ns < 4; ++ns) {
#pragma unroll
        for (int r = 0; r < 8; ++r) {
            int mi = m0 + wave * 16 + r + hi * 8;
            int ni = n0 + ns * 16 + row;
            float v = acc[ns][r];
            if (mode == 0) {
                Cout[(size_t)mi * DMODEL + ni] = v;
            } else {
                int bb = mi >> 11, ss = mi & 2047;   // S = 2048
                int hh = ni >> 6,  dd = ni & 63;     // DH = 64
                Cout[(((size_t)bb * NHEAD + hh) * S_LEN + ss) * DHEAD + dd] = v;
            }
        }
    }
}

// ---------------------------------------------------------------------------
// Postprocess: per (b,h,s) row of DH=64.
// mode (blockIdx.y): 0 = Q (scale by 1/sqrt(DH)); 1/2 = K/V (amax norm +
// fractal compress).  One wave per row, 2 elements per lane.
// ---------------------------------------------------------------------------
__global__ __launch_bounds__(32)
void fractal_post_kernel(const float* __restrict__ Qf, const float* __restrict__ Kf,
                         const float* __restrict__ Vf, const float* __restrict__ kappa,
                         u16* __restrict__ Qo, u16* __restrict__ Ko, u16* __restrict__ Vo) {
    const int rid  = blockIdx.x;          // (b*H + h)*S + s
    const int mode = blockIdx.y;
    const int lane = threadIdx.x;
    const int sIdx = rid & (S_LEN - 1);
    const int h    = (rid >> 11) & (NHEAD - 1);
    const size_t off = (size_t)rid * DHEAD;

    const float* src = (mode == 0) ? Qf : (mode == 1) ? Kf : Vf;
    u16*         dst = (mode == 0) ? Qo : (mode == 1) ? Ko : Vo;

    float x0 = src[off + lane];
    float x1 = src[off + 32 + lane];

    if (mode == 0) {
        dst[off + lane]      = f2b(x0 * 0.125f);   // 1/sqrt(64)
        dst[off + 32 + lane] = f2b(x1 * 0.125f);
        return;
    }
    float amax = fmaxf(fabsf(x0), fabsf(x1));
#pragma unroll
    for (int m = 1; m < 32; m <<= 1) amax = fmaxf(amax, __shfl_xor(amax, m, 32));
    float inv = 1.0f / (amax + 1e-8f);
    float sp  = logf(1.0f + __expf(kappa[h]));                     // softplus
    float age = (float)(S_LEN - 1 - sIdx) / (float)(S_LEN - 1);
    float e   = 1.0f + age * sp;

    float n0 = x0 * inv, n1 = x1 * inv;
    float c0 = copysignf(powf(fmaxf(fabsf(n0), 1e-8f), e), n0);
    float c1 = copysignf(powf(fmaxf(fabsf(n1), 1e-8f), e), n1);
    dst[off + lane]      = f2b(c0);
    dst[off + 32 + lane] = f2b(c1);
}

// ---------------------------------------------------------------------------
// Flash attention (causal), bf16 WMMA, fp32 online softmax.
// Grid: (S/64, B*H). Block: 128 (4 waves); wave w handles q rows [q0b+16w,+16).
// K/V tiles (32 keys x 64 dh, contiguous) TDM-staged, double buffered.
// Output: (b, s, h, dh) bf16 == row-major (B*S, D) for the final projection.
// ---------------------------------------------------------------------------
__global__ __launch_bounds__(128)
void fractal_flash_kernel(const u16* __restrict__ Qb, const u16* __restrict__ Kb,
                          const u16* __restrict__ Vb, u16* __restrict__ Ob) {
    __shared__ u16 Ks[2][32 * 64];
    __shared__ u16 Vs[2][32 * 64];
    __shared__ u16 Ps[4][16 * 32];

    const int tid  = threadIdx.x;
    const int lane = tid & 31;
    const int wave = tid >> 5;
    const int bh = blockIdx.y;
    const int b  = bh >> 4;
    const int h  = bh & 15;
    const size_t base = (size_t)bh * S_LEN * DHEAD;
    const u16* Qg = Qb + base;
    const u16* Kg = Kb + base;
    const u16* Vg = Vb + base;

    const int q0b = blockIdx.x * 64;
    const int q0  = q0b + wave * 16;
    const int row   = lane & 15;
    const int hi    = (lane & 16) ? 1 : 0;
    const int klo_a = hi ? 8 : 0;
    const int kb16  = hi ? 16 : 0;

    // Q A-fragments: two dh-chunks of 32
    v16bf aq[2];
#pragma unroll
    for (int c = 0; c < 2; ++c)
#pragma unroll
        for (int e = 0; e < 16; ++e) {
            int kk = kb16 + e;
            aq[c][e] = us2bf(Qg[(size_t)(q0 + row) * DHEAD + c * 32 + kk]);
        }

    v8f acc[4] = {};
    float mrow[8], lrow[8];
#pragma unroll
    for (int r = 0; r < 8; ++r) { mrow[r] = -3.0e38f; lrow[r] = 0.0f; }

    const int ntiles = (q0b >> 5) + 2;          // keys [0, q0b+64)
    if (tid < 32) {                              // preload tile 0 (contiguous 32x64)
        tdm_load_tile_2d(lds_off_of(Ks[0]), Kg, 2048, 1, 2048);
        tdm_load_tile_2d(lds_off_of(Vs[0]), Vg, 2048, 1, 2048);
    }

    for (int t = 0; t < ntiles; ++t) {
        const int kt0 = t * 32;
        const int cur = t & 1;
        if (tid < 32) {
            if (t + 1 < ntiles) {
                const size_t nxt = (size_t)(kt0 + 32) * DHEAD;
                tdm_load_tile_2d(lds_off_of(Ks[cur ^ 1]), Kg + nxt, 2048, 1, 2048);
                tdm_load_tile_2d(lds_off_of(Vs[cur ^ 1]), Vg + nxt, 2048, 1, 2048);
                __builtin_amdgcn_s_wait_tensorcnt(2);   // tile t landed
            } else {
                __builtin_amdgcn_s_wait_tensorcnt(0);
            }
        }
        __syncthreads();

        // scores: two 16x16 sub-tiles over 32 keys (1/sqrt(dh) pre-folded into Q)
        v8f s[2];
#pragma unroll
        for (int nsub = 0; nsub < 2; ++nsub) {
            v8f st = {};
#pragma unroll
            for (int c = 0; c < 2; ++c) {
                v16bf bfr;
#pragma unroll
                for (int e = 0; e < 16; ++e)
                    bfr[e] = us2bf(Ks[cur][(nsub * 16 + row) * DHEAD + c * 32 + kb16 + e]);
                st = __builtin_amdgcn_wmma_f32_16x16x32_bf16(
                    false, aq[c], false, bfr, (short)0, st, false, false);
            }
            const int key = kt0 + nsub * 16 + row;
#pragma unroll
            for (int r = 0; r < 8; ++r)
                if (key > q0 + r + hi * 8) st[r] = -3.0e38f;   // causal mask
            s[nsub] = st;
        }

        // online softmax per C-row r (row lives across 16 lanes of a half-wave)
#pragma unroll
        for (int r = 0; r < 8; ++r) {
            float mx = fmaxf(s[0][r], s[1][r]);
#pragma unroll
            for (int m = 1; m < 16; m <<= 1) mx = fmaxf(mx, __shfl_xor(mx, m, 32));
            float mnew = fmaxf(mrow[r], mx);
            float resc = __expf(mrow[r] - mnew);
            float p0 = __expf(s[0][r] - mnew);
            float p1 = __expf(s[1][r] - mnew);
            float ps = p0 + p1;
#pragma unroll
            for (int m = 1; m < 16; m <<= 1) ps += __shfl_xor(ps, m, 32);
            lrow[r] = lrow[r] * resc + ps;
            mrow[r] = mnew;
#pragma unroll
            for (int ns = 0; ns < 4; ++ns) acc[ns][r] *= resc;
            const int prow = r + hi * 8;
            Ps[wave][prow * 32 + row]      = f2b(p0);
            Ps[wave][prow * 32 + 16 + row] = f2b(p1);
        }

        // P (A-layout via LDS round trip) x V tile -> accumulate
        v16bf ap;
#pragma unroll
        for (int e = 0; e < 16; ++e) {
            int kk = kb16 + e;
            ap[e] = us2bf(Ps[wave][row * 32 + kk]);
        }
#pragma unroll
        for (int ns = 0; ns < 4; ++ns) {
            v16bf bv;
#pragma unroll
            for (int e = 0; e < 16; ++e)
                bv[e] = us2bf(Vs[cur][(kb16 + e) * DHEAD + ns * 16 + row]);
            acc[ns] = __builtin_amdgcn_wmma_f32_16x16x32_bf16(
                false, ap, false, bv, (short)0, acc[ns], false, false);
        }
        __syncthreads();             // reads of buf `cur` done before reuse
    }

    // epilogue -> (b, s, h, dh) bf16
#pragma unroll
    for (int ns = 0; ns < 4; ++ns)
#pragma unroll
        for (int r = 0; r < 8; ++r) {
            int srow = q0 + r + hi * 8;
            int dh   = ns * 16 + row;
            float v = acc[ns][r] / (lrow[r] + 1e-20f);
            Ob[(((size_t)b * S_LEN + srow) * NHEAD + h) * DHEAD + dh] = f2b(v);
        }
}

// ---------------------------------------------------------------------------
// Launch
// ---------------------------------------------------------------------------
extern "C" void kernel_launch(void* const* d_in, const int* in_sizes, int n_in,
                              void* d_out, int out_size, void* d_ws, size_t ws_size,
                              hipStream_t stream) {
    const float* x     = (const float*)d_in[0];
    const float* Wq    = (const float*)d_in[1];
    const float* Wk    = (const float*)d_in[2];
    const float* Wv    = (const float*)d_in[3];
    const float* Wo    = (const float*)d_in[4];
    const float* kappa = (const float*)d_in[5];
    // d_in[6] (mask) unused: causal mask applied analytically
    float* out = (float*)d_out;

    char* ws = (char*)d_ws;
    const size_t MB = 1024u * 1024u;
    u16*   xb  = (u16*)(ws);                 //  8 MiB bf16 x
    u16*   Wqb = (u16*)(ws + 8 * MB);        //  2 MiB each
    u16*   Wkb = (u16*)(ws + 10 * MB);
    u16*   Wvb = (u16*)(ws + 12 * MB);
    u16*   Wob = (u16*)(ws + 14 * MB);
    float* Qf  = (float*)(ws + 16 * MB);     // 16 MiB fp32 (b,h,s,dh) each
    float* Kf  = (float*)(ws + 32 * MB);
    float* Vf  = (float*)(ws + 48 * MB);
    u16*   Qbh = (u16*)(ws + 64 * MB);       //  8 MiB bf16 each
    u16*   Kbh = (u16*)(ws + 72 * MB);
    u16*   Vbh = (u16*)(ws + 80 * MB);
    u16*   Ab  = (u16*)(ws + 88 * MB);       //  8 MiB bf16 attn out (b,s,d)

    const int nx = MROWS * DMODEL, nw = DMODEL * DMODEL;
    fractal_cvt_kernel<<<(nx + 255) / 256, 256, 0, stream>>>(x, xb, nx);
    fractal_cvt_kernel<<<(nw + 255) / 256, 256, 0, stream>>>(Wq, Wqb, nw);
    fractal_cvt_kernel<<<(nw + 255) / 256, 256, 0, stream>>>(Wk, Wkb, nw);
    fractal_cvt_kernel<<<(nw + 255) / 256, 256, 0, stream>>>(Wv, Wvb, nw);
    fractal_cvt_kernel<<<(nw + 255) / 256, 256, 0, stream>>>(Wo, Wob, nw);

    dim3 g(DMODEL / 64, MROWS / 64);         // (N tiles, M tiles)
    fractal_gemm_kernel<<<g, 128, 0, stream>>>(xb, Wqb, Qf, 1);
    fractal_gemm_kernel<<<g, 128, 0, stream>>>(xb, Wkb, Kf, 1);
    fractal_gemm_kernel<<<g, 128, 0, stream>>>(xb, Wvb, Vf, 1);

    fractal_post_kernel<<<dim3(2 * NHEAD * S_LEN, 3), 32, 0, stream>>>(
        Qf, Kf, Vf, kappa, Qbh, Kbh, Vbh);

    fractal_flash_kernel<<<dim3(S_LEN / 64, 2 * NHEAD), 128, 0, stream>>>(
        Qbh, Kbh, Vbh, Ab);

    fractal_gemm_kernel<<<g, 128, 0, stream>>>(Ab, Wob, out, 0);
}
